// QuantizedLinear_79559974191267
// MI455X (gfx1250) — matrix operand order repaired and checked by reading, not compiled, not run on hardware
//
#include <hip/hip_runtime.h>
#include <hip/hip_bf16.h>

// QuantizedLinear for MI455X (gfx1250, wave32, WMMA).
//
// Reassociated:  out = (x @ R^T) @ codes^T + bias
//   GEMM1: y  [32,4096] = x @ R^T          (reads R, 67 MB, non-temporal)
//   GEMM2: out[32,8192] = y @ codes^T + b  (reads indices, 134 MB, non-temporal)
// Both run on v_wmma_f32_16x16x32_f16 (f16 in, f32 accumulate).
// 4-way split-K inside each workgroup (4 waves/tile, LDS combine).
// GEMM2 dequant uses a 256-entry pair table: one ds_load_b32 yields a packed
// f16 pair (one A-operand VGPR), halving DS ops vs per-element u16 lookups.
//
// Workspace (needs 512 KB):
//   [0      , 256 KB): ws_x  - x swizzled into WMMA B-fragment layout (f16)
//   [256 KB , 512 KB): ws_y  - y^T swizzled into WMMA B-fragment layout (f16)
//
// B-fragment block for (k-tile t of 32, b-plane p of 16):
//   lane l (n = l&15, khalf = l>>4) holds 16 contiguous f16 =
//   B[k = 8*khalf + 0..7, 16 + 8*khalf + 0..7][n]  ->  one 32-byte read per fragment per lane.

typedef __attribute__((ext_vector_type(16))) _Float16 v16h;
typedef __attribute__((ext_vector_type(8)))  _Float16 v8h;
typedef __attribute__((ext_vector_type(8)))  float    v8f;
typedef __attribute__((ext_vector_type(4)))  float    v4f;
typedef __attribute__((ext_vector_type(4)))  int      v4i;
typedef __attribute__((ext_vector_type(8)))  unsigned v8u;

#define DIM   4096
#define OUTF  8192
#define KPART 1024   // DIM / 4 split-K parts

// ---------------------------------------------------------------------------
// Swizzle x [32,4096] f32 -> ws_x B-fragment f16 blocks. 8192 threads.
__global__ __launch_bounds__(256) void prep_x_kernel(
    const float* __restrict__ x, _Float16* __restrict__ wsx) {
  int gid = blockIdx.x * 256 + threadIdx.x;   // 0..8191
  int t  = gid >> 6;                          // k-tile 0..127
  int p  = (gid >> 5) & 1;                    // b-plane
  int l  = gid & 31;                          // lane slot
  int n  = l & 15, kh = l >> 4;
  int b  = p * 16 + n;
  const float* src = x + (size_t)b * DIM + t * 32 + kh * 8;
  v8h lo, hi;
#pragma unroll
  for (int i = 0; i < 8; ++i) lo[i] = (_Float16)src[i];
#pragma unroll
  for (int i = 0; i < 8; ++i) hi[i] = (_Float16)src[16 + i];
  v8h* dst = (v8h*)(wsx + ((size_t)(t * 2 + p) * 32 + l) * 16);
  dst[0] = lo; dst[1] = hi;
}

// ---------------------------------------------------------------------------
// GEMM1: yT[j,b] = sum_d R[j,d] * x[b,d].
// Block = 8 waves = 2 j-chunks (32 rows each) x 4 K-parts.
// Epilogue: sum parts in LDS, transpose, emit ws_y B-fragment layout.
__global__ __launch_bounds__(256) void rot_gemm_kernel(
    const float* __restrict__ R, const _Float16* __restrict__ wsx,
    _Float16* __restrict__ wsy) {
  __shared__ float ysc[2][4][32][33];         // [chunk][part][j_local][b]
  int tid  = threadIdx.x;
  int wave = tid >> 5;
  int l    = tid & 31;
  int cb   = wave >> 2;                       // chunk in block (0..1)
  int part = wave & 3;                        // K part (0..3)
  int chunk = blockIdx.x * 2 + cb;            // 0..127 -> j0 = chunk*32
  int j0 = chunk * 32;
  int m = l & 15, half = l >> 4;
  int d_beg = part * KPART;

  v8f acc[2][2] = {};                         // [j-tile][b-plane]
  for (int d0 = d_beg; d0 < d_beg + KPART; d0 += 32) {
    int t = d0 >> 5;
    v16h a[2];
#pragma unroll
    for (int jt = 0; jt < 2; ++jt) {
      const float* rp = R + (size_t)(j0 + jt * 16 + m) * DIM + d0;
      v4f f0 = __builtin_nontemporal_load((const v4f*)(rp + half * 8));
      v4f f1 = __builtin_nontemporal_load((const v4f*)(rp + half * 8 + 4));
      v4f f2 = __builtin_nontemporal_load((const v4f*)(rp + 16 + half * 8));
      v4f f3 = __builtin_nontemporal_load((const v4f*)(rp + 16 + half * 8 + 4));
#pragma unroll
      for (int i = 0; i < 4; ++i) {
        a[jt][i]      = (_Float16)f0[i];
        a[jt][4 + i]  = (_Float16)f1[i];
        a[jt][8 + i]  = (_Float16)f2[i];
        a[jt][12 + i] = (_Float16)f3[i];
      }
    }
#pragma unroll
    for (int p = 0; p < 2; ++p) {
      v16h bfrag = *(const v16h*)(wsx + ((size_t)(t * 2 + p) * 32 + l) * 16);
#pragma unroll
      for (int jt = 0; jt < 2; ++jt)
        acc[jt][p] = __builtin_amdgcn_wmma_f32_16x16x32_f16(
            false, a[jt], false, bfrag, (short)0, acc[jt][p], false, false);
    }
  }

  // D-tiles -> LDS partials
  int mb = half * 8;
#pragma unroll
  for (int jt = 0; jt < 2; ++jt)
#pragma unroll
    for (int p = 0; p < 2; ++p)
#pragma unroll
      for (int r = 0; r < 8; ++r)
        ysc[cb][part][jt * 16 + mb + r][p * 16 + m] = acc[jt][p][r];
  __syncthreads();

  // Waves 0..3 combine parts and repack to ws_y B-fragment layout.
  if (wave < 4) {
    int cb2 = wave >> 1, p = wave & 1;
    int chunk2 = blockIdx.x * 2 + cb2;
    int n = l & 15, kh = l >> 4;
    v8h lo, hi;
#pragma unroll
    for (int i = 0; i < 8; ++i) {
      float s0 = 0.f, s1 = 0.f;
#pragma unroll
      for (int q = 0; q < 4; ++q) {
        s0 += ysc[cb2][q][kh * 8 + i][p * 16 + n];
        s1 += ysc[cb2][q][16 + kh * 8 + i][p * 16 + n];
      }
      lo[i] = (_Float16)s0;
      hi[i] = (_Float16)s1;
    }
    v8h* dst = (v8h*)(wsy + ((size_t)(chunk2 * 2 + p) * 32 + l) * 16);
    dst[0] = lo; dst[1] = hi;
  }
}

// ---------------------------------------------------------------------------
// GEMM2: out[b,o] = sum_d cb[idx[o,d]] * y[b,d] + bias[o].
// Block = 8 waves = 2 o-tiles (16 rows) x 4 K-parts; LDS combine + bias.
// Dequant: cbs2[(i1<<4)|i0] = packed {f16(cb[i0]) lo, f16(cb[i1]) hi} so one
// ds_load_b32 produces one A-fragment VGPR (consecutive-K f16 pair).
__global__ __launch_bounds__(256) void qlinear_kernel(
    const int* __restrict__ idx, const float* __restrict__ codebook,
    const _Float16* __restrict__ wsy, const float* __restrict__ bias,
    float* __restrict__ out) {
  __shared__ float qsc[2][2][4][16][16];      // [tile][plane][part][m][n]
  __shared__ unsigned cbs2[256];
  int tid = threadIdx.x;
  {
    union { _Float16 h[2]; unsigned u; } pk;
    pk.h[0] = (_Float16)codebook[tid & 15];   // low  half = even-K entry
    pk.h[1] = (_Float16)codebook[tid >> 4];   // high half = odd-K  entry
    cbs2[tid] = pk.u;
  }
  __syncthreads();

  int wave = tid >> 5, l = tid & 31;
  int tb   = wave >> 2;                       // o-tile in block (0..1)
  int part = wave & 3;                        // K part
  int o0 = (blockIdx.x * 2 + tb) * 16;        // 512 o-tiles total
  int m = l & 15, half = l >> 4;
  int d_beg = part * KPART;

  v8f acc[2] = {};
  const int* ip = idx + (size_t)(o0 + m) * DIM;
  for (int d0 = d_beg; d0 < d_beg + KPART; d0 += 32) {
    int t = d0 >> 5;
    v4i i0 = __builtin_nontemporal_load((const v4i*)(ip + d0 + half * 8));
    v4i i1 = __builtin_nontemporal_load((const v4i*)(ip + d0 + half * 8 + 4));
    v4i i2 = __builtin_nontemporal_load((const v4i*)(ip + d0 + 16 + half * 8));
    v4i i3 = __builtin_nontemporal_load((const v4i*)(ip + d0 + 16 + half * 8 + 4));
    v8u au;
    au[0] = cbs2[(i0[0] | (i0[1] << 4)) & 0xFF];
    au[1] = cbs2[(i0[2] | (i0[3] << 4)) & 0xFF];
    au[2] = cbs2[(i1[0] | (i1[1] << 4)) & 0xFF];
    au[3] = cbs2[(i1[2] | (i1[3] << 4)) & 0xFF];
    au[4] = cbs2[(i2[0] | (i2[1] << 4)) & 0xFF];
    au[5] = cbs2[(i2[2] | (i2[3] << 4)) & 0xFF];
    au[6] = cbs2[(i3[0] | (i3[1] << 4)) & 0xFF];
    au[7] = cbs2[(i3[2] | (i3[3] << 4)) & 0xFF];
    v16h a = __builtin_bit_cast(v16h, au);
#pragma unroll
    for (int p = 0; p < 2; ++p) {
      v16h bfrag = *(const v16h*)(wsy + ((size_t)(t * 2 + p) * 32 + l) * 16);
      acc[p] = __builtin_amdgcn_wmma_f32_16x16x32_f16(
          false, a, false, bfrag, (short)0, acc[p], false, false);
    }
  }

  // Partials -> LDS
  int mb = half * 8;
#pragma unroll
  for (int p = 0; p < 2; ++p)
#pragma unroll
    for (int r = 0; r < 8; ++r)
      qsc[tb][p][part][mb + r][m] = acc[p][r];
  __syncthreads();

  // Combine 4 parts + bias, coalesced store: consecutive tid -> consecutive o.
  int o_base = blockIdx.x * 32;
#pragma unroll
  for (int k = 0; k < 4; ++k) {
    int e  = tid + k * 256;                   // 0..1023
    int b  = e >> 5;                          // batch row 0..31
    int ol = e & 31;                          // o within block
    int t2 = ol >> 4, m2 = ol & 15;
    float s = 0.f;
#pragma unroll
    for (int q = 0; q < 4; ++q) s += qsc[t2][b >> 4][q][m2][b & 15];
    out[(size_t)b * OUTF + o_base + ol] = s + bias[o_base + ol];
  }
}

// ---------------------------------------------------------------------------
extern "C" void kernel_launch(void* const* d_in, const int* in_sizes, int n_in,
                              void* d_out, int out_size, void* d_ws, size_t ws_size,
                              hipStream_t stream) {
  const float* x        = (const float*)d_in[0];
  const int*   indices  = (const int*)d_in[1];
  const float* codebook = (const float*)d_in[2];
  const float* R        = (const float*)d_in[3];
  const float* bias     = (const float*)d_in[4];
  float*       out      = (float*)d_out;

  _Float16* wsx = (_Float16*)d_ws;                         // 256 KB
  _Float16* wsy = (_Float16*)((char*)d_ws + 256 * 1024);   // 256 KB

  prep_x_kernel  <<<32,  256, 0, stream>>>(x, wsx);
  rot_gemm_kernel<<<64,  256, 0, stream>>>(R, wsx, wsy);
  qlinear_kernel <<<256, 256, 0, stream>>>(indices, codebook, wsy, bias, out);
}
